// DeformOffsetNet_65893388255309
// MI455X (gfx1250) — matrix operand order repaired and checked by reading, not compiled, not run on hardware
//
#include <hip/hip_runtime.h>

// MI455X / gfx1250 implementation of DeformOffsetNet.
// All 3x3 convs (incl. transposed convs via zero-upsampling) run as
// implicit-GEMM on v_wmma_f32_16x16x32_f16 with fused BN/ReLU/sigmoid epilogue.
// Conv kernel is templated on <CIN, STRIDE> so the K loop fully unrolls and
// all LDS/global addresses fold to immediate offsets.  Each wave computes a
// 2x2 grid of 16x16 tiles (2 co-tiles x 2 pixel-tiles) -> 4 WMMAs per K chunk
// reusing 2 A fragments and 2 B fragments.

typedef __attribute__((ext_vector_type(16))) _Float16 v16h;
typedef __attribute__((ext_vector_type(8)))  float    v8f;
typedef _Float16 h16;

#define BN_EPS 1e-5f

// --------------------------------------------------------------------------
// Implicit-GEMM 3x3 conv.
// K linearization: k = (ky*3+kx)*CIN + ci  (CIN power of two) => within one
// 32-wide K chunk, (ky,kx) is constant and ci contiguous: B fragment is one
// contiguous 32B LDS read at a compile-time immediate offset.
// LDS patch layout: patch[(ry*PWp + cx)*CIN + ci].
// Block: 128 threads stage the patch; waves 0..COUTp/32-1 compute.
// --------------------------------------------------------------------------
template<int CIN, int STRIDE>
__global__ __launch_bounds__(128) void conv_t(
    const h16* __restrict__ X, const h16* __restrict__ Wp,
    const float* __restrict__ scale, const float* __restrict__ bias,
    const h16* __restrict__ res,
    h16* __restrict__ Yh, float* __restrict__ Yf,
    int Hin, int Win, int COUTp, int COUTr, int act,
    int xChTot, int xChOff, int yChTot, int yChOff, int resChTot)
{
  constexpr int NKC  = (CIN * 9) / 32;      // # of 32-wide K chunks (exact)
  constexpr int PW   = 31 * STRIDE + 3;     // patch width for 32 out pixels
  constexpr int PWp  = (PW + 1) & ~1;
  constexpr int SPAN = 3 * PWp;

  const int Ho = Hin / STRIDE, Wo = Win / STRIDE;
  const int lane = threadIdx.x & 31;
  const int wave = threadIdx.x >> 5;
  const int x0 = blockIdx.x * 32;
  const int oy = blockIdx.y;
  const int b  = blockIdx.z;

  __shared__ h16 patch[SPAN * CIN];

  const size_t HWin = (size_t)Hin * Win;
  const int iy0 = oy * STRIDE - 1, ix0 = x0 * STRIDE - 1;
  const h16* Xb = X + (size_t)(b * xChTot + xChOff) * HWin;

  // cooperative, coalesced patch load (zero padded)
  for (int idx = threadIdx.x; idx < SPAN * CIN; idx += 128) {
    int ci = idx / SPAN, pos = idx - ci * SPAN;
    int ry = pos / PWp, cx = pos - ry * PWp;
    int gy = iy0 + ry, gx = ix0 + cx;
    h16 v = (h16)0.f;
    if (gy >= 0 && gy < Hin && gx >= 0 && gx < Win && cx < PW)
      v = Xb[(size_t)ci * HWin + (size_t)gy * Win + gx];
    patch[pos * CIN + ci] = v;
  }
  __syncthreads();

  const int NW = COUTp >> 5;                // compute waves (1 or 2)
  if (wave >= NW) return;

  const int N = lane & 15;                  // pixel within 16-wide tile
  const int kHalf = (lane >> 4) << 4;       // B: lanes 16-31 hold K=16..31
  const int col0 = (N)      * STRIDE * CIN;
  const int col1 = (N + 16) * STRIDE * CIN;

  v8f acc00 = {}, acc01 = {}, acc10 = {}, acc11 = {};
  const h16* wp0 = Wp + ((size_t)((2 * wave) * NKC) * 32 + lane) * 16;
  const h16* wp1 = wp0 + (size_t)NKC * 32 * 16;
  const h16* pb  = patch + kHalf;

  #pragma unroll
  for (int kc = 0; kc < NKC; ++kc) {
    const int kyx    = (kc * 32) / CIN;     // compile-time after unroll
    const int ciBase = (kc * 32) % CIN;
    const int ky = kyx / 3, kx = kyx - ky * 3;
    const int cOff = (ky * PWp + kx) * CIN + ciBase;   // constant
    v16h a0 = *(const v16h*)(wp0 + (size_t)kc * 32 * 16);
    v16h a1 = *(const v16h*)(wp1 + (size_t)kc * 32 * 16);
    v16h b0 = *(const v16h*)&pb[col0 + cOff];
    v16h b1 = *(const v16h*)&pb[col1 + cOff];
    acc00 = __builtin_amdgcn_wmma_f32_16x16x32_f16(false, a0, false, b0, (short)0, acc00, false, false);
    acc01 = __builtin_amdgcn_wmma_f32_16x16x32_f16(false, a0, false, b1, (short)0, acc01, false, false);
    acc10 = __builtin_amdgcn_wmma_f32_16x16x32_f16(false, a1, false, b0, (short)0, acc10, false, false);
    acc11 = __builtin_amdgcn_wmma_f32_16x16x32_f16(false, a1, false, b1, (short)0, acc11, false, false);
  }

  // epilogue: C/D layout -> lane L holds N=L%16, VGPR j holds M = j + (L<16?0:8)
  const int M0 = (lane >> 4) << 3;
  const size_t HWo = (size_t)Ho * Wo;
  const size_t rowBase = (size_t)oy * Wo + x0 + N;

  auto epi = [&](const v8f& a, int coBase, int px) {
    #pragma unroll
    for (int j = 0; j < 8; ++j) {
      int co = coBase + M0 + j;
      if (co >= COUTr) continue;
      float v = a[j];
      if (scale) v = v * scale[co] + bias[co];
      if (res)   v += (float)res[((size_t)(b * resChTot + co)) * HWo + rowBase + px];
      if (act == 1)      v = v > 0.f ? v : 0.f;          // relu
      else if (act == 2) v = 1.f / (1.f + __expf(-v));   // sigmoid
      size_t oi = ((size_t)(b * yChTot + yChOff + co)) * HWo + rowBase + px;
      if (Yh) Yh[oi] = (h16)v;
      if (Yf) Yf[oi] = v;
    }
  };
  epi(acc00, (2 * wave) * 16,     0);
  epi(acc01, (2 * wave) * 16,     16);
  epi(acc10, (2 * wave + 1) * 16, 0);
  epi(acc11, (2 * wave + 1) * 16, 16);
}

// --------------------------------------------------------------------------
// Pack fp32 weights into per-lane WMMA A-fragment order (f16).
// A order used here (contiguous K, same as B): lane L holds M=L%16; slot i -> K = base + i,
// base = (L>=16)*16.  deconv: w[ci][co][2-ky][2-kx].
// --------------------------------------------------------------------------
__global__ void pack_w(const float* __restrict__ W, h16* __restrict__ Wp,
                       int CIN, int ciShift, int COUTr, int deconv, int total)
{
  int idx = blockIdx.x * blockDim.x + threadIdx.x;
  if (idx >= total) return;
  int i    = idx & 15;
  int lane = (idx >> 4) & 31;
  int rest = idx >> 9;
  int NKC  = (CIN * 9) >> 5;
  int kc = rest % NKC;
  int t  = rest / NKC;
  int M = lane & 15;
  int aBase = (lane >> 4) << 4;
  int Klocal = aBase + i;
  int k = kc * 32 + Klocal;
  int ci  = k & (CIN - 1);
  int kyx = k >> ciShift;
  int ky = kyx / 3, kx = kyx - ky * 3;
  int co = t * 16 + M;
  float v = 0.f;
  if (co < COUTr)
    v = deconv ? W[((size_t)ci * COUTr + co) * 9 + (2 - ky) * 3 + (2 - kx)]
               : W[((size_t)co * CIN + ci) * 9 + ky * 3 + kx];
  Wp[idx] = (h16)v;
}

// --------------------------------------------------------------------------
__global__ void bn_prep(const float* __restrict__ g, const float* __restrict__ b,
                        const float* __restrict__ m, const float* __restrict__ v,
                        float* __restrict__ sc, float* __restrict__ bi, int C)
{
  int c = threadIdx.x;
  if (c < C) {
    float s = g[c] * rsqrtf(v[c] + BN_EPS);
    sc[c] = s;
    bi[c] = b[c] - m[c] * s;
  }
}

__global__ void cvt_f32_f16(const float* __restrict__ X, h16* __restrict__ Y,
                            int total, int C, int HW, int yChTot, int yChOff)
{
  int idx = blockIdx.x * blockDim.x + threadIdx.x;
  if (idx >= total) return;
  int pq = idx % HW; int t = idx / HW; int c = t % C; int b = t / C;
  Y[((size_t)(b * yChTot + yChOff + c)) * HW + pq] = (h16)X[idx];
}

// zero-upsample by 2x (lhs_dilation for transposed conv), output contiguous
__global__ void dilate2x(const h16* __restrict__ X, h16* __restrict__ Y,
                         int total, int C, int Hin, int Win, int xChTot, int xChOff)
{
  int idx = blockIdx.x * blockDim.x + threadIdx.x;
  if (idx >= total) return;
  int W2 = Win * 2, H2 = Hin * 2;
  int x = idx % W2; int t = idx / W2; int y = t % H2; t /= H2;
  int c = t % C; int b = t / C;
  h16 v = (h16)0.f;
  if (!((x | y) & 1))
    v = X[((size_t)(b * xChTot + xChOff + c)) * Hin * Win + (size_t)(y >> 1) * Win + (x >> 1)];
  Y[idx] = v;
}

__global__ void copy_base(const float* __restrict__ X, float* __restrict__ dDisp,
                          int total, int HW)
{
  int idx = blockIdx.x * blockDim.x + threadIdx.x;
  if (idx >= total) return;
  int b = idx / HW; int pq = idx - b * HW;
  dDisp[((size_t)b * 7) * HW + pq] = X[idx];   // disp channel 0 = base_disparity
}

// transform sigmoid outputs into final offsets / masks (written to d_out)
__global__ void postoff_k(const float* __restrict__ OFF, float* __restrict__ outOff,
                          float* __restrict__ outMask, int total, int HW,
                          float r, float prer, int part)
{
  int idx = blockIdx.x * blockDim.x + threadIdx.x;
  if (idx >= total) return;
  int pq = idx % HW; int t = idx / HW; int c = t % 27; int b = t / 27;
  float s = OFF[idx];
  if (c < 18) {
    float o = 2.f * s - 1.f;
    if (part) {
      float sg = (o > 0.f) ? 1.f : ((o < 0.f) ? -1.f : 0.f);
      o = o * (r - prer) + sg * prer;
    } else {
      o = o * r;
    }
    outOff[((size_t)(b * 18 + c)) * HW + pq] = o;
  } else {
    outMask[((size_t)(b * 9 + (c - 18))) * HW + pq] = s;
  }
}

// modulated deformable sampling of base_disparity (1 channel, 3x3 kernel)
__global__ void mdcn_k(const float* __restrict__ img, const float* __restrict__ offs,
                       const float* __restrict__ masks, const float* __restrict__ wk,
                       float* __restrict__ disp, int ch, int total, int H, int W)
{
  int idx = blockIdx.x * blockDim.x + threadIdx.x;
  if (idx >= total) return;
  int x = idx % W; int t = idx / W; int y = t % H; int b = t / H;
  const float* im = img + (size_t)b * H * W;
  size_t pix = (size_t)y * W + x;
  float accum = 0.f;
  #pragma unroll
  for (int k = 0; k < 9; ++k) {
    int ky = k / 3, kx = k - ky * 3;
    float ys = (float)(y - 1 + ky) + offs[((size_t)(b * 18 + 2 * k)) * H * W + pix];
    float xs = (float)(x - 1 + kx) + offs[((size_t)(b * 18 + 2 * k + 1)) * H * W + pix];
    float y0f = floorf(ys), x0f = floorf(xs);
    float wy = ys - y0f, wx = xs - x0f;
    int y0 = (int)y0f, x0i = (int)x0f;
    float v00 = 0.f, v01 = 0.f, v10 = 0.f, v11 = 0.f;
    if (y0 >= 0 && y0 < H) {
      if (x0i >= 0     && x0i < W)     v00 = im[(size_t)y0 * W + x0i];
      if (x0i + 1 >= 0 && x0i + 1 < W) v01 = im[(size_t)y0 * W + x0i + 1];
    }
    if (y0 + 1 >= 0 && y0 + 1 < H) {
      if (x0i >= 0     && x0i < W)     v10 = im[(size_t)(y0 + 1) * W + x0i];
      if (x0i + 1 >= 0 && x0i + 1 < W) v11 = im[(size_t)(y0 + 1) * W + x0i + 1];
    }
    float samp = (1.f - wy) * (1.f - wx) * v00 + (1.f - wy) * wx * v01 +
                 wy * (1.f - wx) * v10 + wy * wx * v11;
    accum += samp * masks[((size_t)(b * 9 + k)) * H * W + pix] * wk[k];
  }
  disp[((size_t)(b * 7 + ch)) * H * W + pix] = accum;
}

// ==========================================================================
extern "C" void kernel_launch(void* const* d_in, const int* in_sizes, int n_in,
                              void* d_out, int out_size, void* d_ws, size_t ws_size,
                              hipStream_t stream)
{
  (void)in_sizes; (void)n_in; (void)out_size; (void)ws_size;
  const int Bn = 4, H = 256, Wd = 512; const int HW = H * Wd;
  const int H2 = 128, W2 = 256;
  const int H4 = 64,  W4 = 128;

  // ---- unpack inputs (JAX pytree: dict keys in sorted order) ----
  int p = 0;
  const float* context  = (const float*)d_in[p++];
  const float* baseDisp = (const float*)d_in[p++];
  struct BNp { const float *b, *g, *m, *v; };
  struct CWp { BNp bn; const float* w; };
  auto rdBN = [&]() -> BNp {
    BNp x; x.b = (const float*)d_in[p++]; x.g = (const float*)d_in[p++];
    x.m = (const float*)d_in[p++]; x.v = (const float*)d_in[p++]; return x;
  };
  auto rdCW = [&]() -> CWp { CWp x; x.bn = rdBN(); x.w = (const float*)d_in[p++]; return x; };
  struct Blk { CWp concat; const float* full_dcn; const float* full_off;
               CWp c1, c2, c3, c4, c5, c6;
               const float* part_dcn; const float* part_off; };
  Blk blks[3];
  for (int i = 0; i < 3; ++i) {
    blks[i].concat   = rdCW();
    blks[i].full_dcn = (const float*)d_in[p++];
    blks[i].full_off = (const float*)d_in[p++];
    blks[i].c1 = rdCW(); blks[i].c2 = rdCW(); blks[i].c3 = rdCW();
    blks[i].c4 = rdCW(); blks[i].c5 = rdCW(); blks[i].c6 = rdCW();
    blks[i].part_dcn = (const float*)d_in[p++];
    blks[i].part_off = (const float*)d_in[p++];
  }
  CWp conv0 = rdCW();

  // ---- workspace carve ----
  char* wsb = (char*)d_ws; size_t off = 0;
  auto carve = [&](size_t bytes) -> void* {
    void* r = wsb + off; off = (off + bytes + 255) & ~(size_t)255; return r;
  };
  h16* FBA = (h16*)carve((size_t)Bn * 64 * HW * 2);         // 64ch full-res (ctx|hg)
  h16* FBB = (h16*)carve((size_t)Bn * 64 * HW * 2);
  h16* HB0 = (h16*)carve((size_t)Bn * 64 * H2 * W2 * 2);    // half-res scratch
  h16* HB1 = (h16*)carve((size_t)Bn * 64 * H2 * W2 * 2);    // pre
  h16* HB2 = (h16*)carve((size_t)Bn * 64 * H2 * W2 * 2);    // post
  h16* HB3 = (h16*)carve((size_t)Bn * 64 * H2 * W2 * 2);    // pre_new
  h16* QB0 = (h16*)carve((size_t)Bn * 64 * H4 * W4 * 2);
  h16* QB1 = (h16*)carve((size_t)Bn * 64 * H4 * W4 * 2);
  float* OFF = (float*)carve((size_t)Bn * 27 * HW * 4);     // sigmoid conv output
  h16*   WP  = (h16*)carve((size_t)64 * 576 * 2);           // packed weights (reused)
  float* SC  = (float*)carve(64 * 4);
  float* BI  = (float*)carve(64 * 4);

  float* dout  = (float*)d_out;
  float* dDisp = dout;                                       // [4][7][HW]
  float* dOffs = dout + (size_t)Bn * 7 * HW;                 // 6 x [4][18][HW]
  float* dMask = dOffs + (size_t)6 * Bn * 18 * HW;           // 6 x [4][9][HW]

  auto nb = [&](long total) { return dim3((unsigned)((total + 255) / 256)); };

  auto convL = [&](const float* w, const BNp* bn, int CIN, int COUTp, int COUTr,
                   int deconv, const h16* X, int Hin, int Win, int stride, int act,
                   const h16* res, int resChTot, h16* Yh, float* Yf,
                   int xChTot, int xChOff, int yChTot, int yChOff)
  {
    int ciShift = (CIN == 64) ? 6 : 5;
    int NKC = (CIN * 9) >> 5;
    int tot = (COUTp / 16) * NKC * 32 * 16;
    pack_w<<<nb(tot), 256, 0, stream>>>(w, WP, CIN, ciShift, COUTr, deconv, tot);
    const float *sc = nullptr, *bi = nullptr;
    if (bn) {
      bn_prep<<<1, 64, 0, stream>>>(bn->g, bn->b, bn->m, bn->v, SC, BI, COUTr);
      sc = SC; bi = BI;
    }
    int Wo = Win / stride, Ho = Hin / stride;
    dim3 g(Wo / 32, Ho, Bn), blkDim(128);
    if (CIN == 32 && stride == 1)
      conv_t<32,1><<<g, blkDim, 0, stream>>>(X, WP, sc, bi, res, Yh, Yf,
          Hin, Win, COUTp, COUTr, act, xChTot, xChOff, yChTot, yChOff, resChTot);
    else if (CIN == 32 && stride == 2)
      conv_t<32,2><<<g, blkDim, 0, stream>>>(X, WP, sc, bi, res, Yh, Yf,
          Hin, Win, COUTp, COUTr, act, xChTot, xChOff, yChTot, yChOff, resChTot);
    else if (CIN == 64 && stride == 1)
      conv_t<64,1><<<g, blkDim, 0, stream>>>(X, WP, sc, bi, res, Yh, Yf,
          Hin, Win, COUTp, COUTr, act, xChTot, xChOff, yChTot, yChOff, resChTot);
    else
      conv_t<64,2><<<g, blkDim, 0, stream>>>(X, WP, sc, bi, res, Yh, Yf,
          Hin, Win, COUTp, COUTr, act, xChTot, xChOff, yChTot, yChOff, resChTot);
  };

  // disp channel 0 = base disparity
  copy_base<<<nb((long)Bn * HW), 256, 0, stream>>>(baseDisp, dDisp, Bn * HW, HW);
  // context fp32 -> f16 into FBB ch0-31
  cvt_f32_f16<<<nb((long)Bn * 32 * HW), 256, 0, stream>>>(context, FBB, Bn * 32 * HW, 32, HW, 64, 0);
  // conv0: 32->32, relu, bn  -> ctx in FBA ch0-31
  convL(conv0.w, &conv0.bn, 32, 32, 32, 0, FBB, H, Wd, 1, 1,
        nullptr, 0, FBA, nullptr, 64, 0, 64, 0);

  h16 *cur = FBA, *nxt = FBB, *PRE = HB1, *POST = HB2, *PRENEW = HB3;
  const float radii[3] = {1.f, 2.f, 4.f};

  for (int i = 0; i < 3; ++i) {
    Blk& B_ = blks[i];
    float r = radii[i], prer = i ? radii[i - 1] : 0.f;

    // ---- hourglass ----
    // c1: 32->64 stride2, relu
    convL(B_.c1.w, &B_.c1.bn, 32, 64, 64, 0, cur, H, Wd, 2, 1,
          nullptr, 0, HB0, nullptr, 64, 0, 64, 0);
    // c2: 64->64, bn, (+post_in), relu  -> pre_new
    convL(B_.c2.w, &B_.c2.bn, 64, 64, 64, 0, HB0, H2, W2, 1, 1,
          (i ? POST : nullptr), 64, PRENEW, nullptr, 64, 0, 64, 0);
    // c3: 64->64 stride2, relu
    convL(B_.c3.w, &B_.c3.bn, 64, 64, 64, 0, PRENEW, H2, W2, 2, 1,
          nullptr, 0, QB0, nullptr, 64, 0, 64, 0);
    // c4: 64->64, relu
    convL(B_.c4.w, &B_.c4.bn, 64, 64, 64, 0, QB0, H4, W4, 1, 1,
          nullptr, 0, QB1, nullptr, 64, 0, 64, 0);
    // c5: transposed conv = dilate + flipped conv; bn, +pre(_in), relu -> post
    dilate2x<<<nb((long)Bn * 64 * H2 * W2), 256, 0, stream>>>(QB1, HB0, Bn * 64 * H2 * W2, 64, H4, W4, 64, 0);
    convL(B_.c5.w, &B_.c5.bn, 64, 64, 64, 1, HB0, H2, W2, 1, 1,
          (i ? PRE : PRENEW), 64, POST, nullptr, 64, 0, 64, 0);
    // c6: transposed conv 64->32, bn only -> hg into cur ch32-63
    dilate2x<<<nb((long)Bn * 64 * H * Wd), 256, 0, stream>>>(POST, nxt, Bn * 64 * H * Wd, 64, H2, W2, 64, 0);
    convL(B_.c6.w, &B_.c6.bn, 64, 32, 32, 1, nxt, H, Wd, 1, 0,
          nullptr, 0, cur, nullptr, 64, 0, 64, 32);

    // ---- concat conv: [ctx|hg] 64 -> 32, relu -> new ctx in nxt ch0-31 ----
    convL(B_.concat.w, &B_.concat.bn, 64, 32, 32, 0, cur, H, Wd, 1, 1,
          nullptr, 0, nxt, nullptr, 64, 0, 64, 0);

    // ---- part offsets ----
    convL(B_.part_off, nullptr, 32, 32, 27, 0, nxt, H, Wd, 1, 2,
          nullptr, 0, nullptr, OFF, 64, 0, 27, 0);
    {
      int oi = 2 * i;
      float* oOff = dOffs + (size_t)oi * Bn * 18 * HW;
      float* oMsk = dMask + (size_t)oi * Bn * 9 * HW;
      postoff_k<<<nb((long)Bn * 27 * HW), 256, 0, stream>>>(OFF, oOff, oMsk, Bn * 27 * HW, HW, r, prer, 1);
      mdcn_k<<<nb((long)Bn * HW), 256, 0, stream>>>(baseDisp, oOff, oMsk, B_.part_dcn, dDisp, 1 + 2 * i, Bn * HW, H, Wd);
    }
    // ---- full offsets ----
    convL(B_.full_off, nullptr, 32, 32, 27, 0, nxt, H, Wd, 1, 2,
          nullptr, 0, nullptr, OFF, 64, 0, 27, 0);
    {
      int oi = 2 * i + 1;
      float* oOff = dOffs + (size_t)oi * Bn * 18 * HW;
      float* oMsk = dMask + (size_t)oi * Bn * 9 * HW;
      postoff_k<<<nb((long)Bn * 27 * HW), 256, 0, stream>>>(OFF, oOff, oMsk, Bn * 27 * HW, HW, r, prer, 0);
      mdcn_k<<<nb((long)Bn * HW), 256, 0, stream>>>(baseDisp, oOff, oMsk, B_.full_dcn, dDisp, 2 + 2 * i, Bn * HW, H, Wd);
    }

    { h16* t = PRE; PRE = PRENEW; PRENEW = t; }
    { h16* t = cur; cur = nxt; nxt = t; }
  }
}
